// BinaryLinear_2_4_14448269984134
// MI455X (gfx1250) — compile-verified, matrix operand order checked by reading, not run on hardware
//
#include <hip/hip_runtime.h>

// ---------------------------------------------------------------------------
// C = binary_quant(2:4-masked W) @ X on gfx1250 (MI455X), f32 in/out.
//
// Two-phase pipeline (when ws_size >= 96MB):
//   1) quant_w_tiled : W f32 -> WB bf16 {0,1}, WMMA A-fragment-tiled layout
//   2) split_x_tiled : X f32 -> (Xhi, Xlo) bf16, WMMA B-fragment-tiled layout
//   3) gemm_bf16_frag: double-buffered LDS GEMM; per K-step all 8 fragments
//      are loaded into distinct registers first, then 16 WMMAs issue
//      back-to-back (no ds-wait between quads). 2 WMMAs (hi+lo) per logical
//      MAC tile -> near-f32 accuracy at bf16 WMMA speed.
// Fallback (small ws): fused single-kernel version (round-1).
//
// Fragment-tiled layout: operand tile (16 rows x 32 k) = 1024B chunk;
// lane L owns 32B = 16 bf16 in WMMA element order:
//   element e -> K = e + 8*(e>=8) + 8*(L>=16),  row = L%16.
// ---------------------------------------------------------------------------

typedef __attribute__((ext_vector_type(16))) __bf16 v16bf;
typedef __attribute__((ext_vector_type(8)))  __bf16 v8bf;
typedef __attribute__((ext_vector_type(8)))  float  v8f;

#define MD 4096
#define KD 4096
#define ND 4096
#define KT32 (KD / 32)      // 128 k-tiles

__device__ __forceinline__ unsigned short bf16_rne_bits(float f) {
    unsigned u = __builtin_bit_cast(unsigned, f);
    unsigned r = u + 0x7FFFu + ((u >> 16) & 1u);
    return (unsigned short)(r >> 16);
}

__device__ __forceinline__ void bf16_split(float f, unsigned short& hb, unsigned short& lb) {
    unsigned u  = __builtin_bit_cast(unsigned, f);
    unsigned rh = u + 0x7FFFu + ((u >> 16) & 1u);
    hb = (unsigned short)(rh >> 16);
    float hf = __builtin_bit_cast(float, (unsigned)hb << 16);
    lb = bf16_rne_bits(f - hf);
}

// 2:4 top-2-|.| mask + binary quant, jax.lax.top_k tie-breaking (earlier idx wins).
__device__ __forceinline__ uint2 bin24_pack(float4 w4) {
    const float a0 = fabsf(w4.x), a1 = fabsf(w4.y);
    const float a2 = fabsf(w4.z), a3 = fabsf(w4.w);
    const int r0 = (a1 >  a0) + (a2 >  a0) + (a3 >  a0);
    const int r1 = (a0 >= a1) + (a2 >  a1) + (a3 >  a1);
    const int r2 = (a0 >= a2) + (a1 >= a2) + (a3 >  a2);
    const int r3 = (a0 >= a3) + (a1 >= a3) + (a2 >= a3);
    const unsigned b0 = (r0 < 2 && w4.x > 0.f) ? 0x3F80u : 0u;
    const unsigned b1 = (r1 < 2 && w4.y > 0.f) ? 0x3F80u : 0u;
    const unsigned b2 = (r2 < 2 && w4.z > 0.f) ? 0x3F80u : 0u;
    const unsigned b3 = (r3 < 2 && w4.w > 0.f) ? 0x3F80u : 0u;
    return make_uint2(b0 | (b1 << 16), b2 | (b3 << 16));
}

__device__ __forceinline__ v16bf load_frag(const __bf16* p) {
    const v8bf lo = *(const v8bf*)p;
    const v8bf hi = *(const v8bf*)(p + 8);
    return __builtin_shufflevector(lo, hi,
        0, 1, 2, 3, 4, 5, 6, 7, 8, 9, 10, 11, 12, 13, 14, 15);
}

__device__ __forceinline__ v8f wmma_bf16(v16bf a, v16bf b, v8f c) {
    return __builtin_amdgcn_wmma_f32_16x16x32_bf16(
        false, a, false, b, (short)0, c, false, false);
}

// ---------------- phase 1a: W -> binary 2:4 bf16, fragment-tiled ----------------

__global__ __launch_bounds__(256)
void quant_w_tiled(const float* __restrict__ W, __bf16* __restrict__ WBt) {
    __shared__ __bf16 sW[128][32];

    const int t  = threadIdx.x;
    const int r8 = t >> 3;
    const int k4 = (t & 7) * 4;
    const int mbase = blockIdx.x * 128;
    const int kbase = blockIdx.y * 32;

#pragma unroll
    for (int g = 0; g < 4; ++g) {
        const int row = r8 + 32 * g;
        const float4 w4 = *(const float4*)(W + (size_t)(mbase + row) * KD
                                             + (size_t)(kbase + k4));
        *reinterpret_cast<uint2*>(&sW[row][k4]) = bin24_pack(w4);
    }
    __syncthreads();

    const int mt = t >> 5, L = t & 31;
    const int kb = (L >> 4) * 8, mr = L & 15;
    const __bf16* srow = &sW[mt * 16 + mr][0];
    const uint4 c0 = *reinterpret_cast<const uint4*>(srow + kb);
    const uint4 c1 = *reinterpret_cast<const uint4*>(srow + kb + 16);
    __bf16* dst = WBt + ((size_t)(blockIdx.x * 8 + mt) * KT32 + blockIdx.y) * 512
                      + (size_t)L * 16;
    *reinterpret_cast<uint4*>(dst)     = c0;
    *reinterpret_cast<uint4*>(dst + 8) = c1;
}

// ---------------- phase 1b: X -> (hi, lo) bf16, fragment-tiled ----------------

__global__ __launch_bounds__(256)
void split_x_tiled(const float* __restrict__ X,
                   __bf16* __restrict__ Xht, __bf16* __restrict__ Xlt) {
    __shared__ unsigned sH[128][20];
    __shared__ unsigned sL[128][20];

    const int t  = threadIdx.x;
    const int kp = t >> 4;
    const int n8 = (t & 15) * 8;
    const int kbase = blockIdx.y * 32;
    const int nbase = blockIdx.x * 128;

    const float* xr0 = X + (size_t)(kbase + 2 * kp) * ND + (size_t)(nbase + n8);
    const float* xr1 = xr0 + ND;
    const float4 a0 = ((const float4*)xr0)[0], a1 = ((const float4*)xr0)[1];
    const float4 b0 = ((const float4*)xr1)[0], b1 = ((const float4*)xr1)[1];
    const float v0[8] = {a0.x, a0.y, a0.z, a0.w, a1.x, a1.y, a1.z, a1.w};
    const float v1[8] = {b0.x, b0.y, b0.z, b0.w, b1.x, b1.y, b1.z, b1.w};

#pragma unroll
    for (int j = 0; j < 8; ++j) {
        unsigned short h0, l0, h1, l1;
        bf16_split(v0[j], h0, l0);
        bf16_split(v1[j], h1, l1);
        sH[n8 + j][kp] = (unsigned)h0 | ((unsigned)h1 << 16);
        sL[n8 + j][kp] = (unsigned)l0 | ((unsigned)l1 << 16);
    }
    __syncthreads();

    const int nt = t >> 5, L = t & 31;
    const int hq = (L >> 4) * 4, mr = L & 15;
    const unsigned* hrow = &sH[nt * 16 + mr][0];
    const unsigned* lrow = &sL[nt * 16 + mr][0];
    const uint4 h0v = *reinterpret_cast<const uint4*>(hrow + hq);
    const uint4 h1v = *reinterpret_cast<const uint4*>(hrow + hq + 8);
    const uint4 l0v = *reinterpret_cast<const uint4*>(lrow + hq);
    const uint4 l1v = *reinterpret_cast<const uint4*>(lrow + hq + 8);

    const size_t off = ((size_t)(blockIdx.x * 8 + nt) * KT32 + blockIdx.y) * 512
                       + (size_t)L * 16;
    *reinterpret_cast<uint4*>(Xht + off)     = h0v;
    *reinterpret_cast<uint4*>(Xht + off + 8) = h1v;
    *reinterpret_cast<uint4*>(Xlt + off)     = l0v;
    *reinterpret_cast<uint4*>(Xlt + off + 8) = l1v;
}

// ---------------- phase 2: WMMA GEMM from fragment-tiled operands ----------------
// Block 256 thr = 8 waves, tile 128(M) x 128(N), BK=32, double-buffered LDS.
// Wave (2x4 grid) owns 64(M) x 32(N): 16 WMMAs per K-step, issued as one burst.

__global__ __launch_bounds__(256)
void gemm_bf16_frag(const __bf16* __restrict__ WBt,
                    const __bf16* __restrict__ Xht,
                    const __bf16* __restrict__ Xlt,
                    float* __restrict__ C) {
    __shared__ __bf16 sA[2][4096];
    __shared__ __bf16 sH[2][4096];
    __shared__ __bf16 sL[2][4096];

    const int t    = threadIdx.x;
    const int L    = t & 31;
    const int wave = t >> 5;
    const int wm   = wave >> 2;
    const int wn   = wave & 3;

    const int bm0 = blockIdx.y * 8;
    const int bn0 = blockIdx.x * 8;

    const int sOff = wave * 512 + L * 16;
    const __bf16* pA = WBt + (size_t)(bm0 + wave) * KT32 * 512 + (size_t)L * 16;
    const __bf16* pH = Xht + (size_t)(bn0 + wave) * KT32 * 512 + (size_t)L * 16;
    const __bf16* pL = Xlt + (size_t)(bn0 + wave) * KT32 * 512 + (size_t)L * 16;

    v8f acc[4][2] = {};

    uint4 ra0 = *reinterpret_cast<const uint4*>(pA);
    uint4 ra1 = *reinterpret_cast<const uint4*>(pA + 8);
    uint4 rh0 = *reinterpret_cast<const uint4*>(pH);
    uint4 rh1 = *reinterpret_cast<const uint4*>(pH + 8);
    uint4 rl0 = *reinterpret_cast<const uint4*>(pL);
    uint4 rl1 = *reinterpret_cast<const uint4*>(pL + 8);

    int buf = 0;
    for (int kt = 0; kt < KT32; ++kt) {
        __bf16* dA = &sA[buf][sOff];
        __bf16* dH = &sH[buf][sOff];
        __bf16* dL = &sL[buf][sOff];
        *reinterpret_cast<uint4*>(dA)     = ra0;
        *reinterpret_cast<uint4*>(dA + 8) = ra1;
        *reinterpret_cast<uint4*>(dH)     = rh0;
        *reinterpret_cast<uint4*>(dH + 8) = rh1;
        *reinterpret_cast<uint4*>(dL)     = rl0;
        *reinterpret_cast<uint4*>(dL + 8) = rl1;

        // next K-step's global loads in flight during the WMMA burst below
        if (kt + 1 < KT32) {
            pA += 512; pH += 512; pL += 512;
            ra0 = *reinterpret_cast<const uint4*>(pA);
            ra1 = *reinterpret_cast<const uint4*>(pA + 8);
            rh0 = *reinterpret_cast<const uint4*>(pH);
            rh1 = *reinterpret_cast<const uint4*>(pH + 8);
            rl0 = *reinterpret_cast<const uint4*>(pL);
            rl1 = *reinterpret_cast<const uint4*>(pL + 8);
        }
        __syncthreads();   // single barrier per iter (double buffer)

        // load ALL 8 fragments into distinct registers first ...
        v16bf Af[4], Bhf[2], Blf[2];
#pragma unroll
        for (int i = 0; i < 4; ++i)
            Af[i] = load_frag(&sA[buf][(wm * 4 + i) * 512 + L * 16]);
#pragma unroll
        for (int j = 0; j < 2; ++j) {
            const int nt = wn * 2 + j;
            Bhf[j] = load_frag(&sH[buf][nt * 512 + L * 16]);
            Blf[j] = load_frag(&sL[buf][nt * 512 + L * 16]);
        }

        // ... then 16 WMMAs back-to-back (no LDS waits between quads)
#pragma unroll
        for (int j = 0; j < 2; ++j) {
#pragma unroll
            for (int i = 0; i < 4; ++i) {
                acc[i][j] = wmma_bf16(Af[i], Bhf[j], acc[i][j]);   // A * x_hi
                acc[i][j] = wmma_bf16(Af[i], Blf[j], acc[i][j]);   // A * x_lo
            }
        }
        buf ^= 1;
    }

    const int hh = L >> 4, mr = L & 15;
#pragma unroll
    for (int i = 0; i < 4; ++i) {
        const int m0 = (bm0 + wm * 4 + i) * 16 + hh * 8;
#pragma unroll
        for (int j = 0; j < 2; ++j) {
            const int n0 = (bn0 + wn * 2 + j) * 16 + mr;
#pragma unroll
            for (int r = 0; r < 8; ++r)
                C[(size_t)(m0 + r) * ND + n0] = acc[i][j][r];
        }
    }
}

// ---------------- fallback: fused single kernel (round-1, known-good) ----------------

#define FBM 32
#define FBN 64
#define FBK 32
#define FXS 40

__global__ __launch_bounds__(256)
void bin24_wmma_fused(const float* __restrict__ X,
                      const float* __restrict__ W,
                      float* __restrict__ C) {
    __shared__ __bf16 fA[FBM][FBK];
    __shared__ __bf16 fH[FBN][FXS];
    __shared__ __bf16 fL[FBN][FXS];

    const int tid  = threadIdx.x;
    const int lane = tid & 31;
    const int wave = tid >> 5;
    const int wm   = wave >> 2;
    const int wn   = wave & 3;
    const int hh   = lane >> 4;
    const int mr   = lane & 15;
    const int mblk = blockIdx.y * FBM;
    const int nblk = blockIdx.x * FBN;
    const int a_row = tid >> 3, a_k4 = (tid & 7) * 4;
    const int x_k   = tid >> 3, x_n8 = (tid & 7) * 8;
    const int kb = hh * 8;

    v8f acc = {};
    for (int kt = 0; kt < KD / FBK; ++kt) {
        const int k0 = kt * FBK;
        {
            const float4 w4 = *(const float4*)(W + (size_t)(mblk + a_row) * KD
                                                 + (size_t)(k0 + a_k4));
            *reinterpret_cast<uint2*>(&fA[a_row][a_k4]) = bin24_pack(w4);
        }
        {
            const float* xp = X + (size_t)(k0 + x_k) * ND + (size_t)(nblk + x_n8);
            const float4 q0 = ((const float4*)xp)[0];
            const float4 q1 = ((const float4*)xp)[1];
            const float vals[8] = {q0.x, q0.y, q0.z, q0.w, q1.x, q1.y, q1.z, q1.w};
#pragma unroll
            for (int j = 0; j < 8; ++j) {
                unsigned short hb, lb;
                bf16_split(vals[j], hb, lb);
                fH[x_n8 + j][x_k] = __builtin_bit_cast(__bf16, hb);
                fL[x_n8 + j][x_k] = __builtin_bit_cast(__bf16, lb);
            }
        }
        __syncthreads();

        const __bf16* arow = &fA[wm * 16 + mr][0];
        const v16bf Afr = __builtin_shufflevector(
            *(const v8bf*)(arow + kb), *(const v8bf*)(arow + kb + 16),
            0,1,2,3,4,5,6,7,8,9,10,11,12,13,14,15);
        const __bf16* hrow = &fH[wn * 16 + mr][0];
        const v16bf Bh = __builtin_shufflevector(
            *(const v8bf*)(hrow + kb), *(const v8bf*)(hrow + kb + 16),
            0,1,2,3,4,5,6,7,8,9,10,11,12,13,14,15);
        const __bf16* lrow = &fL[wn * 16 + mr][0];
        const v16bf Bl = __builtin_shufflevector(
            *(const v8bf*)(lrow + kb), *(const v8bf*)(lrow + kb + 16),
            0,1,2,3,4,5,6,7,8,9,10,11,12,13,14,15);

        acc = wmma_bf16(Afr, Bh, acc);
        acc = wmma_bf16(Afr, Bl, acc);
        __syncthreads();
    }

    const int m0 = mblk + wm * 16 + hh * 8;
    const int n0 = nblk + wn * 16 + mr;
#pragma unroll
    for (int r = 0; r < 8; ++r)
        C[(size_t)(m0 + r) * ND + n0] = acc[r];
}

// ---------------- launch ----------------

extern "C" void kernel_launch(void* const* d_in, const int* in_sizes, int n_in,
                              void* d_out, int out_size, void* d_ws, size_t ws_size,
                              hipStream_t stream) {
    (void)in_sizes; (void)n_in; (void)out_size;
    const float* X = (const float*)d_in[0];   // (K, N) f32
    const float* W = (const float*)d_in[1];   // (M, K) f32
    float* C = (float*)d_out;                 // (M, N) f32

    const size_t elems = (size_t)MD * KD;
    const size_t need  = elems * 2 * 3;       // WB + Xhi + Xlo bf16 = 96MB

    if (d_ws != nullptr && ws_size >= need) {
        __bf16* WBt = (__bf16*)d_ws;
        __bf16* Xht = WBt + elems;
        __bf16* Xlt = Xht + elems;
        quant_w_tiled<<<dim3(MD / 128, KD / 32), 256, 0, stream>>>(W, WBt);
        split_x_tiled<<<dim3(ND / 128, KD / 32), 256, 0, stream>>>(X, Xht, Xlt);
        gemm_bf16_frag<<<dim3(ND / 128, MD / 128), 256, 0, stream>>>(WBt, Xht, Xlt, C);
    } else {
        bin24_wmma_fused<<<dim3(ND / FBN, MD / FBM), 256, 0, stream>>>(X, W, C);
    }
}
